// HSS_21002390077572
// MI455X (gfx1250) — compile-verified, matrix operand order checked by reading, not run on hardware
//
#include <hip/hip_runtime.h>
#include <cstdint>

#define NBATCH 4
#define HDIM   128
#define WDIM   128
#define LSEQ   (HDIM * WDIM)        // 16384
#define NTOK   (NBATCH * LSEQ)      // 65536
#define DMODEL 128
#define DINNER 256
#define DSTATE 16
#define DTRANK 8
#define C40    40                   // DT_RANK + 2*D_STATE
#define C48    48                   // padded column count for x_proj
#define NCH    128                  // scan chunks per sequence
#define CHS    (LSEQ / NCH)         // 128 steps per chunk

typedef __attribute__((ext_vector_type(16))) __bf16 v16bf;
typedef __attribute__((ext_vector_type(8)))  __bf16 v8bf;
typedef __attribute__((ext_vector_type(8)))  float  v8f;

// ---------------- WMMA fragment loaders (wave32, 16x16x32 bf16) -------------
// A-matrix 16x32 (16-bit): lane holds row m = lane&15; lane-half selects K
// sub-range: VGPR0..3 = K[kg*8 .. kg*8+7], VGPR4..7 = K[16+kg*8 ..], kg=lane>>4.
// B-matrix 32x16 is the mirrored layout with n = lane&15 (column per lane).

// f32 source (network input x only): load + convert.
__device__ __forceinline__ v16bf load_tile_f32cvt(const float* __restrict__ base,
                                                  int ld, int lane) {
  const int r  = lane & 15;
  const int kg = (lane >> 4) << 3;      // 0 or 8
  const float* p = base + (size_t)r * ld + kg;
  v16bf f;
#pragma unroll
  for (int j = 0; j < 8; ++j) {
    f[j]     = (__bf16)p[j];
    f[8 + j] = (__bf16)p[16 + j];
  }
  return f;
}

// Native bf16 source: two b128 loads, no VALU converts.
__device__ __forceinline__ v16bf load_tile_bf16n(const __bf16* __restrict__ base,
                                                 int ld, int lane) {
  const int r  = lane & 15;
  const int kg = (lane >> 4) << 3;
  const __bf16* p = base + (size_t)r * ld + kg;
  const v8bf lo = *(const v8bf*)p;
  const v8bf hi = *(const v8bf*)(p + 16);
  return __builtin_shufflevector(lo, hi, 0, 1, 2, 3, 4, 5, 6, 7,
                                 8, 9, 10, 11, 12, 13, 14, 15);
}

// Hilbert-gathered bf16 A fragment (rows are gathered tokens of one image).
__device__ __forceinline__ v16bf load_tileA_hil(const __bf16* __restrict__ xcb,
                                                const int* __restrict__ hil,
                                                size_t btok, int l0, int kb, int lane) {
  const int r  = lane & 15;
  const int kg = (lane >> 4) << 3;
  const __bf16* p = xcb + (btok + (size_t)hil[l0 + r]) * DINNER + kb + kg;
  const v8bf lo = *(const v8bf*)p;
  const v8bf hi = *(const v8bf*)(p + 16);
  return __builtin_shufflevector(lo, hi, 0, 1, 2, 3, 4, 5, 6, 7,
                                 8, 9, 10, 11, 12, 13, 14, 15);
}

// ---------------- 0) one-shot weight conversions ----------------------------
__global__ void k_cvt(const float* __restrict__ src, __bf16* __restrict__ dst, int n) {
  const int i = blockIdx.x * blockDim.x + threadIdx.x;
  if (i < n) dst[i] = (__bf16)src[i];
}

// x_proj_w (40x256) -> padded (48x256) bf16, zero rows 40..47 (kills guards).
__global__ void k_cvt_xpw(const float* __restrict__ src, __bf16* __restrict__ dst) {
  const int i = blockIdx.x * blockDim.x + threadIdx.x;   // C48*DINNER threads
  const int rr = i >> 8, cc = i & (DINNER - 1);
  dst[i] = (rr < C40) ? (__bf16)src[rr * DINNER + cc] : (__bf16)0.f;
}

// ---------------- 1) in_proj GEMM: xz[t, 0:512] = x[t, :] . W^T ------------
// One wave = one 16-token row tile x 4 column tiles (32 acc VGPRs: no spills);
// 8 waves/block cover all 32 column tiles. A fragment loaded once per K-step
// and reused for 4 WMMAs; A re-reads across waves hit L2 (x fits easily).
__global__ void k_inproj(const float* __restrict__ x, const __bf16* __restrict__ w,
                         float* __restrict__ xg, float* __restrict__ z) {
  const int lane = threadIdx.x & 31;
  const int wid  = threadIdx.x >> 5;        // 0..7
  const int row0 = blockIdx.x << 4;         // token tile
  const int ct0  = wid << 2;                // first of 4 column tiles (32 total)
  const v8f zero = {};
  v8f acc[4];
#pragma unroll
  for (int t = 0; t < 4; ++t) acc[t] = zero;
#pragma unroll
  for (int kb = 0; kb < DMODEL; kb += 32) {
    const v16bf a = load_tile_f32cvt(x + (size_t)row0 * DMODEL + kb, DMODEL, lane);
#pragma unroll
    for (int t = 0; t < 4; ++t) {
      const int col0 = (ct0 + t) << 4;
      const v16bf bf = load_tile_bf16n(w + (size_t)col0 * DMODEL + kb, DMODEL, lane);
      acc[t] = __builtin_amdgcn_wmma_f32_16x16x32_bf16(false, a, false, bf,
                                                       (short)0, acc[t], false, false);
    }
  }
  const int n  = lane & 15;
  const int mh = (lane >> 4) << 3;          // C/D: lanes>=16 hold rows m+8
#pragma unroll
  for (int t = 0; t < 4; ++t) {
    const int col0 = (ct0 + t) << 4;
#pragma unroll
    for (int r = 0; r < 8; ++r) {
      const int tok = row0 + mh + r;
      const int d   = col0 + n;
      if (d < DINNER) xg[(size_t)tok * DINNER + d] = acc[t][r];
      else            z [(size_t)tok * DINNER + (d - DINNER)] = acc[t][r];
    }
  }
}

// ---------------- 2) depthwise 3x3 conv (SAME) + bias + SiLU ----------------
// Emits xc in f32 (scan input u) and bf16 (x_proj GEMM A operand).
__global__ void k_conv(const float* __restrict__ xg, const float* __restrict__ cw,
                       const float* __restrict__ cb, float* __restrict__ xc,
                       __bf16* __restrict__ xcb) {
  const int i = blockIdx.x * blockDim.x + threadIdx.x;  // NTOK*DINNER threads
  const int d = i & (DINNER - 1);
  const int t = i >> 8;
  const int p = t & (LSEQ - 1);
  const int b = t >> 14;
  const int h = p >> 7, ww = p & (WDIM - 1);
  float acc = cb[d];
#pragma unroll
  for (int di = -1; di <= 1; ++di) {
#pragma unroll
    for (int dj = -1; dj <= 1; ++dj) {
      const int hh = h + di, wj = ww + dj;
      if (hh >= 0 && hh < HDIM && wj >= 0 && wj < WDIM)
        acc += xg[((size_t)b * LSEQ + hh * WDIM + wj) * DINNER + d] *
               cw[d * 9 + (di + 1) * 3 + (dj + 1)];
    }
  }
  const float s = acc / (1.f + __expf(-acc));           // SiLU
  xc [(size_t)t * DINNER + d] = s;
  xcb[(size_t)t * DINNER + d] = (__bf16)s;
}

// ---------------- 3) x_proj GEMM with fused Hilbert gather ------------------
// xdbl[t, c] = sum_d xc[b, hil[l], d] * w[c, d]; weights padded to 48 cols.
// One wave = one row tile x all 3 column tiles; gathered A loaded once per K-step.
__global__ void k_xproj(const __bf16* __restrict__ xcb, const int* __restrict__ hil,
                        const __bf16* __restrict__ w, float* __restrict__ xdbl) {
  const int lane  = threadIdx.x & 31;
  const int wid   = threadIdx.x >> 5;                   // 0..3
  const int trow0 = ((blockIdx.x << 2) + wid) << 4;
  const int b     = trow0 >> 14;
  const int l0    = trow0 & (LSEQ - 1);
  const v8f zero = {};
  v8f acc[3];
#pragma unroll
  for (int t = 0; t < 3; ++t) acc[t] = zero;
#pragma unroll
  for (int kb = 0; kb < DINNER; kb += 32) {
    const v16bf a = load_tileA_hil(xcb, hil, (size_t)b * LSEQ, l0, kb, lane);
#pragma unroll
    for (int t = 0; t < 3; ++t) {
      const int col0 = t << 4;
      const v16bf bf = load_tile_bf16n(w + (size_t)col0 * DINNER + kb, DINNER, lane);
      acc[t] = __builtin_amdgcn_wmma_f32_16x16x32_bf16(false, a, false, bf,
                                                       (short)0, acc[t], false, false);
    }
  }
  const int n  = lane & 15;
  const int mh = (lane >> 4) << 3;
#pragma unroll
  for (int t = 0; t < 3; ++t) {
    const int c = (t << 4) + n;
    if (c < C40) {
#pragma unroll
      for (int r = 0; r < 8; ++r)
        xdbl[(size_t)(trow0 + mh + r) * C40 + c] = acc[t][r];
    }
  }
}

// ---------------- 4) chunked selective scan ---------------------------------
// Recurrence per (b,d,n): h <- h*exp(delta*A) + delta*u*B_l ; y = h.C_l
// Phase 1: per-chunk (prod a, partial state with h_in = 0)
__global__ void k_scan1(const float* __restrict__ xc, const int* __restrict__ hil,
                        const float* __restrict__ xdbl, const float* __restrict__ dtw,
                        const float* __restrict__ dtb, const float* __restrict__ alog,
                        float* __restrict__ F, float* __restrict__ P) {
  const int b = blockIdx.x / NCH;
  const int c = blockIdx.x % NCH;
  const int d = threadIdx.x;
  float A[DSTATE];
#pragma unroll
  for (int n = 0; n < DSTATE; ++n) A[n] = -__expf(alog[d * DSTATE + n]);
  float dw[DTRANK];
#pragma unroll
  for (int r = 0; r < DTRANK; ++r) dw[r] = dtw[d * DTRANK + r];
  const float db = dtb[d];
  float h[DSTATE], Pp[DSTATE];
#pragma unroll
  for (int n = 0; n < DSTATE; ++n) { h[n] = 0.f; Pp[n] = 1.f; }
  __shared__ float sx[C40];
  const size_t btok = (size_t)b * LSEQ;
  for (int s = 0; s < CHS; ++s) {
    const int l = c * CHS + s;
    __syncthreads();
    if (threadIdx.x < C40) sx[threadIdx.x] = xdbl[(btok + l) * C40 + threadIdx.x];
    __syncthreads();
    const float u = xc[(btok + hil[l]) * DINNER + d];
    float dts = db;
#pragma unroll
    for (int r = 0; r < DTRANK; ++r) dts += sx[r] * dw[r];
    const float delta = (dts > 20.f) ? dts : log1pf(__expf(dts));
    const float du = delta * u;
#pragma unroll
    for (int n = 0; n < DSTATE; ++n) {
      const float a = __expf(delta * A[n]);
      h[n]  = h[n] * a + du * sx[8 + n];
      Pp[n] *= a;
    }
  }
  const size_t base = (((size_t)b * NCH + c) * DINNER + d) * DSTATE;
#pragma unroll
  for (int n = 0; n < DSTATE; ++n) { F[base + n] = h[n]; P[base + n] = Pp[n]; }
}

// Phase 2: serial carry over the 128 chunks (tiny); F becomes incoming state.
__global__ void k_scan2(float* __restrict__ F, const float* __restrict__ P) {
  const int i = blockIdx.x * blockDim.x + threadIdx.x;
  if (i >= NBATCH * DINNER * DSTATE) return;
  const int b  = i / (DINNER * DSTATE);
  const int dn = i % (DINNER * DSTATE);
  float hc = 0.f;
  for (int c = 0; c < NCH; ++c) {
    const size_t idx = ((size_t)b * NCH + c) * (DINNER * DSTATE) + dn;
    const float Fi = F[idx], Pi = P[idx];
    F[idx] = hc;                       // incoming state for chunk c
    hc = hc * Pi + Fi;
  }
}

// Phase 3: replay each chunk with correct h_in, emit y (+ D*u skip).
__global__ void k_scan3(const float* __restrict__ xc, const int* __restrict__ hil,
                        const float* __restrict__ xdbl, const float* __restrict__ dtw,
                        const float* __restrict__ dtb, const float* __restrict__ alog,
                        const float* __restrict__ Dv, const float* __restrict__ F,
                        float* __restrict__ ys) {
  const int b = blockIdx.x / NCH;
  const int c = blockIdx.x % NCH;
  const int d = threadIdx.x;
  float A[DSTATE];
#pragma unroll
  for (int n = 0; n < DSTATE; ++n) A[n] = -__expf(alog[d * DSTATE + n]);
  float dw[DTRANK];
#pragma unroll
  for (int r = 0; r < DTRANK; ++r) dw[r] = dtw[d * DTRANK + r];
  const float db = dtb[d];
  const float Dd = Dv[d];
  float h[DSTATE];
  const size_t fbase = (((size_t)b * NCH + c) * DINNER + d) * DSTATE;
#pragma unroll
  for (int n = 0; n < DSTATE; ++n) h[n] = F[fbase + n];
  __shared__ float sx[C40];
  const size_t btok = (size_t)b * LSEQ;
  for (int s = 0; s < CHS; ++s) {
    const int l = c * CHS + s;
    __syncthreads();
    if (threadIdx.x < C40) sx[threadIdx.x] = xdbl[(btok + l) * C40 + threadIdx.x];
    __syncthreads();
    const float u = xc[(btok + hil[l]) * DINNER + d];
    float dts = db;
#pragma unroll
    for (int r = 0; r < DTRANK; ++r) dts += sx[r] * dw[r];
    const float delta = (dts > 20.f) ? dts : log1pf(__expf(dts));
    const float du = delta * u;
    float y = 0.f;
#pragma unroll
    for (int n = 0; n < DSTATE; ++n) {
      const float a = __expf(delta * A[n]);
      h[n] = h[n] * a + du * sx[8 + n];
      y += h[n] * sx[24 + n];
    }
    ys[(btok + l) * DINNER + d] = y + Dd * u;
  }
}

// ---------------- 5) inverse-Hilbert + LayerNorm + SiLU gate ----------------
// One wave32 per spatial token; each lane handles 8 channels. Emits bf16
// directly (it is the A operand of the out_proj WMMA).
__global__ void k_gate(const float* __restrict__ ys, const float* __restrict__ z,
                       const int* __restrict__ inv, const float* __restrict__ gam,
                       const float* __restrict__ bet, __bf16* __restrict__ yg) {
  const int gw   = (blockIdx.x * blockDim.x + threadIdx.x) >> 5;
  const int lane = threadIdx.x & 31;
  if (gw >= NTOK) return;
  const int b = gw >> 14;
  const int p = gw & (LSEQ - 1);
  const size_t src = ((size_t)b * LSEQ + inv[p]) * DINNER;
  float v[8];
  float s1 = 0.f, s2 = 0.f;
#pragma unroll
  for (int j = 0; j < 8; ++j) {
    const float y = ys[src + lane * 8 + j];
    v[j] = y; s1 += y; s2 += y * y;
  }
#pragma unroll
  for (int off = 16; off > 0; off >>= 1) {
    s1 += __shfl_xor(s1, off, 32);
    s2 += __shfl_xor(s2, off, 32);
  }
  const float mu  = s1 * (1.f / DINNER);
  const float var = s2 * (1.f / DINNER) - mu * mu;
  const float rin = rsqrtf(var + 1e-5f);
  const size_t dst = (size_t)gw * DINNER;
#pragma unroll
  for (int j = 0; j < 8; ++j) {
    const int dd = lane * 8 + j;
    const float y  = (v[j] - mu) * rin * gam[dd] + bet[dd];
    const float zz = z[dst + dd];
    yg[dst + dd] = (__bf16)(y * (zz / (1.f + __expf(-zz))));
  }
}

// ---------------- 6) out_proj GEMM: out[t, m] = yg[t, :] . W^T --------------
// One wave = one row tile x 4 column tiles (no spills); 2 waves per row tile
// cover N=128. Both operands native bf16.
__global__ void k_outproj(const __bf16* __restrict__ yg, const __bf16* __restrict__ w,
                          float* __restrict__ out) {
  const int lane = threadIdx.x & 31;
  const int wid  = threadIdx.x >> 5;        // 0..7
  const int row0 = ((blockIdx.x << 2) + (wid >> 1)) << 4;
  const int ct0  = (wid & 1) << 2;          // col tiles [ct0, ct0+4)
  const v8f zero = {};
  v8f acc[4];
#pragma unroll
  for (int t = 0; t < 4; ++t) acc[t] = zero;
#pragma unroll
  for (int kb = 0; kb < DINNER; kb += 32) {
    const v16bf a = load_tile_bf16n(yg + (size_t)row0 * DINNER + kb, DINNER, lane);
#pragma unroll
    for (int t = 0; t < 4; ++t) {
      const int col0 = (ct0 + t) << 4;
      const v16bf bf = load_tile_bf16n(w + (size_t)col0 * DINNER + kb, DINNER, lane);
      acc[t] = __builtin_amdgcn_wmma_f32_16x16x32_bf16(false, a, false, bf,
                                                       (short)0, acc[t], false, false);
    }
  }
  const int n  = lane & 15;
  const int mh = (lane >> 4) << 3;
#pragma unroll
  for (int t = 0; t < 4; ++t) {
#pragma unroll
    for (int r = 0; r < 8; ++r)
      out[(size_t)(row0 + mh + r) * DMODEL + ((ct0 + t) << 4) + n] = acc[t][r];
  }
}

// ---------------------------------------------------------------------------
extern "C" void kernel_launch(void* const* d_in, const int* in_sizes, int n_in,
                              void* d_out, int out_size, void* d_ws, size_t ws_size,
                              hipStream_t stream) {
  const float* x         = (const float*)d_in[0];
  const float* in_proj_w = (const float*)d_in[1];
  const float* conv_w    = (const float*)d_in[2];
  const float* conv_b    = (const float*)d_in[3];
  const float* x_proj_w  = (const float*)d_in[4];
  const float* dt_proj_w = (const float*)d_in[5];
  const float* dt_proj_b = (const float*)d_in[6];
  const float* A_logs    = (const float*)d_in[7];
  const float* Ds        = (const float*)d_in[8];
  const float* ln_gamma  = (const float*)d_in[9];
  const float* ln_beta   = (const float*)d_in[10];
  const float* out_proj_w= (const float*)d_in[11];
  const int*   hil       = (const int*)d_in[12];
  const int*   inv_hil   = (const int*)d_in[13];
  (void)in_sizes; (void)n_in; (void)out_size; (void)ws_size;

  // Workspace carve-up: f32 region (~296 MB) then bf16 region (~67 MB).
  float* ws = (float*)d_ws;
  const size_t N_ID = (size_t)NTOK * DINNER;                  // 16.78M elements
  float* XG   = ws;                                           // conv input (xg)
  float* Z    = XG + N_ID;                                    // gate branch
  float* XC   = Z + N_ID;                                     // conv+silu output (f32)
  float* YS   = XC + N_ID;                                    // scan output (hilbert order)
  float* XDBL = YS + N_ID;                                    // NTOK*40
  float* F    = XDBL + (size_t)NTOK * C40;                    // B*NCH*DINNER*16
  float* P    = F + (size_t)NBATCH * NCH * DINNER * DSTATE;   // same size
  __bf16* bws = (__bf16*)(P + (size_t)NBATCH * NCH * DINNER * DSTATE);
  __bf16* XCB  = bws;                                         // bf16 conv output
  __bf16* YGB  = XCB + N_ID;                                  // bf16 gated output
  __bf16* WIN  = YGB + N_ID;                                  // bf16 in_proj_w (512x128)
  __bf16* WXP  = WIN + (size_t)(2 * DINNER) * DMODEL;         // bf16 padded x_proj_w (48x256)
  __bf16* WOUT = WXP + (size_t)C48 * DINNER;                  // bf16 out_proj_w (128x256)

  // One-shot weight conversions (tiny; L2-resident afterwards).
  k_cvt<<<(2 * DINNER * DMODEL + 255) / 256, 256, 0, stream>>>(in_proj_w, WIN,
                                                               2 * DINNER * DMODEL);
  k_cvt_xpw<<<(C48 * DINNER + 255) / 256, 256, 0, stream>>>(x_proj_w, WXP);
  k_cvt<<<(DMODEL * DINNER + 255) / 256, 256, 0, stream>>>(out_proj_w, WOUT,
                                                           DMODEL * DINNER);

  k_inproj<<<NTOK / 16, 256, 0, stream>>>(x, WIN, XG, Z);
  k_conv<<<(int)(N_ID / 256), 256, 0, stream>>>(XG, conv_w, conv_b, XC, XCB);
  k_xproj<<<NTOK / 64, 128, 0, stream>>>(XCB, hil, WXP, XDBL);
  k_scan1<<<NBATCH * NCH, DINNER, 0, stream>>>(XC, hil, XDBL, dt_proj_w, dt_proj_b,
                                               A_logs, F, P);
  const int nth2 = NBATCH * DINNER * DSTATE;
  k_scan2<<<(nth2 + 255) / 256, 256, 0, stream>>>(F, P);
  k_scan3<<<NBATCH * NCH, DINNER, 0, stream>>>(XC, hil, XDBL, dt_proj_w, dt_proj_b,
                                               A_logs, Ds, F, YS);
  k_gate<<<NTOK / 8, 256, 0, stream>>>(YS, Z, inv_hil, ln_gamma, ln_beta, YGB);
  k_outproj<<<NTOK / 64, 256, 0, stream>>>(YGB, WOUT, (float*)d_out);
}
